// DotProductAttention_4028679324270
// MI455X (gfx1250) — matrix operand order repaired and checked by reading, not compiled, not run on hardware
//
#include <hip/hip_runtime.h>

// ---------------------------------------------------------------------------
// Flash attention forward, CDNA5 (gfx1250), wave32, WMMA f16.
// B=32, S=2048, D=128, fp32 in/out.
// Block = 256 threads = 8 waves; each wave owns 32 query rows (2 row-tiles:
// every K/V B-fragment read from LDS feeds 2 WMMAs). Double-buffered K/V
// tiles in LDS with register staging, one barrier per KV step.
// QK^T accumulates f32 (softmax precision); O accumulates f16 C/D to cut
// register pressure (peak ~300 VGPRs -> scheduler can pipeline ds loads).
// ---------------------------------------------------------------------------

typedef __attribute__((ext_vector_type(16))) _Float16 v16h;
typedef __attribute__((ext_vector_type(8)))  _Float16 v8h;
typedef __attribute__((ext_vector_type(8)))  float    v8f;

#define B_    32
#define S_    2048
#define D_    128
#define WAVES 8
#define MROWS 32                   // query rows per wave (2 x 16)
#define MTILE (WAVES * MROWS)      // 256 query rows per block
#define NTILE 64                   // kv positions per step
#define NSTEP (S_ / NTILE)
#define KPAD  136                  // K tile row stride in halves (128 + 8)
#define VPAD  72                   // V^T tile row stride in halves (64 + 8)
#define PPAD  72                   // P tile row stride in halves (64 + 8)

// 1/sqrt(128) * log2(e): softmax kept in exp2 domain
#define SCALE_LOG2E 0.1275186f

union FragU { v16h v; uint4 u[2]; };

static __device__ __forceinline__ unsigned pkh(float a, float b) {
    union { _Float16 h[2]; unsigned u; } z;
    z.h[0] = (_Float16)a; z.h[1] = (_Float16)b;
    return z.u;
}

__global__ __launch_bounds__(256, 1)
void fa_fwd(const float* __restrict__ Q, const float* __restrict__ K,
            const float* __restrict__ V, float* __restrict__ O)
{
    __shared__ _Float16 Kl[2][NTILE * KPAD];      // K tiles, row-major [kv][d]
    __shared__ _Float16 Vt[2][D_ * VPAD];         // V tiles, transposed [d][kv]
    __shared__ _Float16 Pl[WAVES * MROWS * PPAD]; // per-wave P staging [32][64]

    const int t    = threadIdx.x;
    const int wave = t >> 5;
    const int lane = t & 31;
    const int lr   = lane & 15;
    const int g    = lane >> 4;

    const int batch = blockIdx.x >> 3;            // 8 q-tiles per batch
    const int q0    = (blockIdx.x & 7) * MTILE;

    const float* Qb = Q + (size_t)batch * S_ * D_;
    const float* Kb = K + (size_t)batch * S_ * D_;
    const float* Vb = V + (size_t)batch * S_ * D_;
    float*       Ob = O + (size_t)batch * S_ * D_;

    // ---- staging thread mapping for cooperative K/V tile loads ----
    const int dcol = (t & 31) * 4;    // 0..124
    const int kr   = t >> 5;          // K rows kr + 8j   (j = 0..7)
    const int vr   = (t >> 5) * 2;    // V row pairs vr + 16j, +1 (j = 0..3)

    float4 kreg[8];
    float4 vreg[4][2];

    auto stage_load = [&](int s) {
#pragma unroll
        for (int j = 0; j < 8; ++j)
            kreg[j] = *(const float4*)(Kb + (size_t)(s + kr + 8 * j) * D_ + dcol);
#pragma unroll
        for (int j = 0; j < 4; ++j) {
            vreg[j][0] = *(const float4*)(Vb + (size_t)(s + vr + 16 * j) * D_ + dcol);
            vreg[j][1] = *(const float4*)(Vb + (size_t)(s + vr + 16 * j + 1) * D_ + dcol);
        }
    };
    auto stage_store = [&](int buf) {
#pragma unroll
        for (int j = 0; j < 8; ++j)
            *(uint2*)&Kl[buf][(kr + 8 * j) * KPAD + dcol] =
                make_uint2(pkh(kreg[j].x, kreg[j].y), pkh(kreg[j].z, kreg[j].w));
#pragma unroll
        for (int j = 0; j < 4; ++j) {
            const float* a = (const float*)&vreg[j][0];
            const float* b = (const float*)&vreg[j][1];
#pragma unroll
            for (int i = 0; i < 4; ++i)   // pack kv-row pair -> one b32 store
                *(unsigned*)&Vt[buf][(dcol + i) * VPAD + vr + 16 * j] = pkh(a[i], b[i]);
        }
    };

    // ---- Q: 2 row-tiles x 4 k-chunks of A-fragments, resident all kernel ----
    v16h qf[2][4];
#pragma unroll
    for (int rt = 0; rt < 2; ++rt) {
        const int qrow = q0 + wave * MROWS + rt * 16 + lr;
#pragma unroll
        for (int kf = 0; kf < 4; ++kf) {
            const float* src = Qb + (size_t)qrow * D_ + kf * 32 + g * 8;
            float tmpv[16];
            *(float4*)&tmpv[0]  = *(const float4*)(src + 0);
            *(float4*)&tmpv[4]  = *(const float4*)(src + 4);
            *(float4*)&tmpv[8]  = *(const float4*)(src + 16);
            *(float4*)&tmpv[12] = *(const float4*)(src + 20);
            v16h f;
#pragma unroll
            for (int i = 0; i < 16; ++i) f[i] = (_Float16)tmpv[i];
            qf[rt][kf] = f;
        }
    }

    // ---- running state (f16 C/D layout: element e <-> row e + 8g) ----
    v8h o_acc[2][8];
#pragma unroll
    for (int rt = 0; rt < 2; ++rt)
#pragma unroll
        for (int dt = 0; dt < 8; ++dt)
#pragma unroll
            for (int e = 0; e < 8; ++e) o_acc[rt][dt][e] = (_Float16)0.0f;

    float mstate[2][8], lstate[2][8];
#pragma unroll
    for (int rt = 0; rt < 2; ++rt)
#pragma unroll
        for (int e = 0; e < 8; ++e) { mstate[rt][e] = -3.0e38f; lstate[rt][e] = 0.0f; }

    // ---- prologue: tile 0 into buffer 0 ----
    stage_load(0);
    stage_store(0);
    __syncthreads();

#pragma unroll 1
    for (int it = 0; it < NSTEP; ++it) {
        const int cur = it & 1;
        const bool have_next = (it + 1) < NSTEP;

        // warm L2 two tiles ahead (global_prefetch_b8); no VGPR cost
        if (it + 2 < NSTEP) {
            __builtin_prefetch(Kb + (size_t)(it + 2) * NTILE * D_ + t * 32, 0, 3);
            __builtin_prefetch(Vb + (size_t)(it + 2) * NTILE * D_ + t * 32, 0, 3);
        }

        const _Float16* Kc = &Kl[cur][0];
        const _Float16* Vc = &Vt[cur][0];

        // ---- S = Q K^T ----
        // Stream order i = kf*4 + nt: dependent accumulator updates are 8 WMMAs
        // apart; each B-fragment is loaded one iteration ahead of its use.
        v8f acc[2][4];
#pragma unroll
        for (int rt = 0; rt < 2; ++rt)
#pragma unroll
            for (int nt = 0; nt < 4; ++nt)
#pragma unroll
                for (int i = 0; i < 8; ++i) acc[rt][nt][i] = 0.0f;

        {
            FragU bcur, bnxt;
            const _Float16* p0 = &Kc[(0 * 16 + lr) * KPAD + 0 * 32 + g * 8];
            bcur.u[0] = *(const uint4*)(p0);
            bcur.u[1] = *(const uint4*)(p0 + 16);
#pragma unroll
            for (int i = 0; i < 16; ++i) {
                if (i + 1 < 16) {
                    const int nt1 = (i + 1) & 3, kf1 = (i + 1) >> 2;
                    const _Float16* p =
                        &Kc[(nt1 * 16 + lr) * KPAD + kf1 * 32 + g * 8];
                    bnxt.u[0] = *(const uint4*)(p);
                    bnxt.u[1] = *(const uint4*)(p + 16);
                }
                const int nt = i & 3, kf = i >> 2;
                acc[0][nt] = __builtin_amdgcn_wmma_f32_16x16x32_f16(
                    false, qf[0][kf], false, bcur.v, (short)0, acc[0][nt], false, false);
                acc[1][nt] = __builtin_amdgcn_wmma_f32_16x16x32_f16(
                    false, qf[1][kf], false, bcur.v, (short)0, acc[1][nt], false, false);
                bcur = bnxt;
            }
        }

        // ---- online softmax (per row-tile; rows are wave-private) ----
#pragma unroll
        for (int rt = 0; rt < 2; ++rt) {
#pragma unroll
            for (int nt = 0; nt < 4; ++nt)
#pragma unroll
                for (int i = 0; i < 8; ++i) acc[rt][nt][i] *= SCALE_LOG2E;

            float alpha[8];
#pragma unroll
            for (int e = 0; e < 8; ++e) {
                float m = fmaxf(fmaxf(acc[rt][0][e], acc[rt][1][e]),
                                fmaxf(acc[rt][2][e], acc[rt][3][e]));
#pragma unroll
                for (int msk = 1; msk <= 8; msk <<= 1)
                    m = fmaxf(m, __shfl_xor(m, msk, 32));  // within 16-lane half
                const float mn = fmaxf(mstate[rt][e], m);
                alpha[e]       = __builtin_amdgcn_exp2f(mstate[rt][e] - mn);
                mstate[rt][e]  = mn;
                lstate[rt][e] *= alpha[e];
            }
            _Float16 ah[8];
#pragma unroll
            for (int e = 0; e < 8; ++e) ah[e] = (_Float16)alpha[e];
#pragma unroll
            for (int dt = 0; dt < 8; ++dt)      // packed f16 rescale of O
#pragma unroll
                for (int e = 0; e < 8; ++e) o_acc[rt][dt][e] *= ah[e];

            float rsum[8];
#pragma unroll
            for (int e = 0; e < 8; ++e) rsum[e] = 0.0f;
#pragma unroll
            for (int nt = 0; nt < 4; ++nt) {
#pragma unroll
                for (int e = 0; e < 8; ++e) {
                    float p = __builtin_amdgcn_exp2f(acc[rt][nt][e] - mstate[rt][e]);
                    rsum[e] += p;
                    // f32 C layout: row = e + 8g; stage f16 for PV A-fragments
                    Pl[(wave * MROWS + rt * 16 + e + 8 * g) * PPAD + nt * 16 + lr] =
                        (_Float16)p;
                }
            }
#pragma unroll
            for (int e = 0; e < 8; ++e) {
#pragma unroll
                for (int msk = 1; msk <= 8; msk <<= 1)
                    rsum[e] += __shfl_xor(rsum[e], msk, 32);
                lstate[rt][e] += rsum[e];
            }
        }

        // per-wave LDS ops are in-order; block compiler reordering only
        __builtin_amdgcn_wave_barrier();

        // issue next tile's global loads now: staging regs live only through
        // the PV phase; PV compute + earlier prefetch hide the latency
        if (have_next) stage_load((it + 1) * NTILE);

        // ---- O += P V (f16 accumulate) ----
        v16h pf[2][2];
#pragma unroll
        for (int rt = 0; rt < 2; ++rt)
#pragma unroll
            for (int kf2 = 0; kf2 < 2; ++kf2) {
                const _Float16* pp =
                    &Pl[(wave * MROWS + rt * 16 + lr) * PPAD + kf2 * 32 + g * 8];
                FragU f;
                f.u[0] = *(const uint4*)(pp);
                f.u[1] = *(const uint4*)(pp + 16);
                pf[rt][kf2] = f.v;
            }
        // Stream order j = kf2*8 + dt: dependent accumulator updates are 16
        // WMMAs apart; V B-fragments loaded one iteration ahead.
        {
            FragU bcur, bnxt;
            const _Float16* p0 = &Vc[(0 * 16 + lr) * VPAD + 0 * 32 + g * 8];
            bcur.u[0] = *(const uint4*)(p0);
            bcur.u[1] = *(const uint4*)(p0 + 16);
#pragma unroll
            for (int j = 0; j < 16; ++j) {
                if (j + 1 < 16) {
                    const int dt1 = (j + 1) & 7, kf1 = (j + 1) >> 3;
                    const _Float16* p =
                        &Vc[(dt1 * 16 + lr) * VPAD + kf1 * 32 + g * 8];
                    bnxt.u[0] = *(const uint4*)(p);
                    bnxt.u[1] = *(const uint4*)(p + 16);
                }
                const int dt = j & 7, kf2 = j >> 3;
                o_acc[0][dt] = __builtin_amdgcn_wmma_f16_16x16x32_f16(
                    false, pf[0][kf2], false, bcur.v, (short)0, o_acc[0][dt], false, false);
                o_acc[1][dt] = __builtin_amdgcn_wmma_f16_16x16x32_f16(
                    false, pf[1][kf2], false, bcur.v, (short)0, o_acc[1][dt], false, false);
                bcur = bnxt;
            }
        }

        // convert + store next tile into the other buffer, then one barrier
        if (have_next) stage_store(cur ^ 1);
        __syncthreads();
    }

    // ---- epilogue: O /= l, scatter (f16 C layout: row = e + 8g) ----
#pragma unroll
    for (int rt = 0; rt < 2; ++rt) {
        float inv[8];
#pragma unroll
        for (int e = 0; e < 8; ++e) inv[e] = 1.0f / lstate[rt][e];
#pragma unroll
        for (int dt = 0; dt < 8; ++dt) {
#pragma unroll
            for (int e = 0; e < 8; ++e) {
                const int row = q0 + wave * MROWS + rt * 16 + e + 8 * g;
                Ob[(size_t)row * D_ + dt * 16 + lr] =
                    (float)o_acc[rt][dt][e] * inv[e];
            }
        }
    }
}

extern "C" void kernel_launch(void* const* d_in, const int* in_sizes, int n_in,
                              void* d_out, int out_size, void* d_ws, size_t ws_size,
                              hipStream_t stream) {
    (void)in_sizes; (void)n_in; (void)out_size; (void)d_ws; (void)ws_size;
    const float* q = (const float*)d_in[0];
    const float* k = (const float*)d_in[1];
    const float* v = (const float*)d_in[2];
    float*       o = (float*)d_out;

    dim3 grid(B_ * (S_ / MTILE));   // 32 batches x 8 query tiles = 256 blocks
    dim3 block(256);                // 8 wave32s
    hipLaunchKernelGGL(fa_fwd, grid, block, 0, stream, q, k, v, o);
}
